// PointCrossAttention_76802605187308
// MI455X (gfx1250) — compile-verified, matrix operand order checked by reading, not run on hardware
//
#include <hip/hip_runtime.h>
#include <hip/hip_bf16.h>
#include <math.h>

// ---------------------------------------------------------------------------
// Model constants (from the reference)
// ---------------------------------------------------------------------------
#define BATCH      2
#define PC_SIZE    32768
#define TOTAL_PTS  65536      // PC_SIZE + SHARP_SIZE
#define IN_GRP     2048       // IN_R == IN_S
#define NS_GRP     512        // NS_R == NS_S
#define N_DATA     4096       // per batch: 2*IN_GRP
#define N_QUERY    1024       // per batch: 2*NS_GRP
#define WIDTH      1024
#define HEADS      16
#define HD         64
#define KPAD       64         // IN_DIM=54 padded to 64
#define M_Q        (BATCH * N_QUERY)   // 2048
#define M_D        (BATCH * N_DATA)    // 8192

typedef __attribute__((ext_vector_type(16))) _Float16 v16h;
typedef __attribute__((ext_vector_type(8)))  _Float16 v8h;
typedef __attribute__((ext_vector_type(8)))  float    v8f;
typedef int v4i __attribute__((vector_size(16)));

#define AS1 __attribute__((address_space(1)))
#define AS3 __attribute__((address_space(3)))

// gfx1250 async global->LDS copy (guarded; falls back to vector LDS store)
#if defined(__HIP_DEVICE_COMPILE__) && __has_builtin(__builtin_amdgcn_global_load_async_to_lds_b128)
#define HAS_ASYNC_LDS 1
#else
#define HAS_ASYNC_LDS 0
#endif

__device__ inline void async_wait0() {
#if defined(__HIP_DEVICE_COMPILE__)
#if __has_builtin(__builtin_amdgcn_s_wait_asynccnt)
  __builtin_amdgcn_s_wait_asynccnt(0);
#else
  asm volatile("s_wait_asynccnt 0" ::: "memory");
#endif
#endif
}

#if HAS_ASYNC_LDS
__device__ inline void async_copy16(const _Float16* gp, _Float16* lp) {
  // addrspacecast (generic -> AS1 / AS3) then bitcast to the vector pointee
  AS1 v4i* gv = (AS1 v4i*)(AS1 _Float16*)gp;
  AS3 v4i* lv = (AS3 v4i*)(AS3 _Float16*)lp;
  __builtin_amdgcn_global_load_async_to_lds_b128(gv, lv, 0, 0);
}
#endif

// ---------------------------------------------------------------------------
// WMMA helpers (CDNA5 wave32, 16x16x32 f16 -> f32)
// ---------------------------------------------------------------------------
__device__ inline v8f wmma_f16(v16h a, v16h b, v8f c) {
  return __builtin_amdgcn_wmma_f32_16x16x32_f16(
      /*neg_a=*/false, a, /*neg_b=*/false, b,
      /*c_mod=*/(short)0, c, /*reuse_a=*/false, /*reuse_b=*/false);
}

// Fragment loader for the 16-bit A-operand layout (16x32, K-contiguous rows):
//   lanes 0-15 : row = lane,    halves 0-7 = K 0..7,  halves 8-15 = K 16..23
//   lanes 16-31: row = lane-16, halves 0-7 = K 8..15, halves 8-15 = K 24..31
// B tiles are staged N-major so the same loader serves both operands.
__device__ inline v16h load_frag(const _Float16* base, int row0, int ld, int lane) {
  const int lr = lane & 15;
  const int hi = (lane >> 4) << 3;  // 0 or 8
  const _Float16* p = base + (size_t)(row0 + lr) * ld + hi;
  v16h f;
#pragma unroll
  for (int j = 0; j < 8; ++j) { f[j] = p[j]; f[j + 8] = p[j + 16]; }
  return f;
}

__device__ inline float gelu_exact(float x) {
  return 0.5f * x * (1.0f + erff(x * 0.70710678118654752f));
}

// ---------------------------------------------------------------------------
// 1) FPS + gather (one block per (batch, group)); sequential 512-step scan.
// ---------------------------------------------------------------------------
__global__ __launch_bounds__(256) void fps_gather_k(
    const float* __restrict__ pc, const float* __restrict__ feats,
    const int* __restrict__ permR, const int* __restrict__ permS,
    float* __restrict__ inpPts, float* __restrict__ qPts,
    float* __restrict__ inpFts, float* __restrict__ qFts) {
  const int b = blockIdx.x >> 1, g = blockIdx.x & 1;
  const int* perm = g ? permS : permR;
  const size_t srcBase = ((size_t)b * TOTAL_PTS + (g ? PC_SIZE : 0)) * 3;
  const float* P = pc + srcBase;
  const float* F = feats + srcBase;
  float* ip  = inpPts + ((size_t)b * N_DATA  + (size_t)g * IN_GRP) * 3;
  float* ifs = inpFts + ((size_t)b * N_DATA  + (size_t)g * IN_GRP) * 3;
  float* qp  = qPts   + ((size_t)b * N_QUERY + (size_t)g * NS_GRP) * 3;
  float* qf  = qFts   + ((size_t)b * N_QUERY + (size_t)g * NS_GRP) * 3;

  __shared__ float Px[IN_GRP], Py[IN_GRP], Pz[IN_GRP];
  __shared__ float Fx[IN_GRP], Fy[IN_GRP], Fz[IN_GRP];
  __shared__ float Dd[IN_GRP];
  __shared__ float rv[256];
  __shared__ int   ri[256];
  __shared__ int   selIdx[NS_GRP];

  const int tid = threadIdx.x;
  for (int j = tid; j < IN_GRP; j += 256) {
    const int pj = perm[j];
    const float x = P[(size_t)pj * 3 + 0];
    const float y = P[(size_t)pj * 3 + 1];
    const float z = P[(size_t)pj * 3 + 2];
    Px[j] = x; Py[j] = y; Pz[j] = z;
    Fx[j] = F[(size_t)pj * 3 + 0];
    Fy[j] = F[(size_t)pj * 3 + 1];
    Fz[j] = F[(size_t)pj * 3 + 2];
    Dd[j] = 1e30f;
    ip[j * 3 + 0] = x; ip[j * 3 + 1] = y; ip[j * 3 + 2] = z;
    ifs[j * 3 + 0] = Fx[j]; ifs[j * 3 + 1] = Fy[j]; ifs[j * 3 + 2] = Fz[j];
  }
  __syncthreads();

  int far = 0;
  for (int s = 0; s < NS_GRP; ++s) {
    if (tid == 0) selIdx[s] = far;
    const float cx = Px[far], cy = Py[far], cz = Pz[far];
    float bv = -1.0f; int bi = 0x7fffffff;
    for (int j = tid; j < IN_GRP; j += 256) {
      const float dx = Px[j] - cx, dy = Py[j] - cy, dz = Pz[j] - cz;
      const float dist = sqrtf(dx * dx + dy * dy + dz * dz);
      const float dn = fminf(Dd[j], dist);
      Dd[j] = dn;
      if (dn > bv) { bv = dn; bi = j; }
    }
    rv[tid] = bv; ri[tid] = bi;
    __syncthreads();
    for (int off = 128; off > 0; off >>= 1) {
      if (tid < off) {
        const float ov = rv[tid + off]; const int oi = ri[tid + off];
        if (ov > rv[tid] || (ov == rv[tid] && oi < ri[tid])) { rv[tid] = ov; ri[tid] = oi; }
      }
      __syncthreads();
    }
    far = ri[0];
    __syncthreads();
  }

  for (int s = tid; s < NS_GRP; s += 256) {
    const int j = selIdx[s];
    qp[s * 3 + 0] = Px[j]; qp[s * 3 + 1] = Py[j]; qp[s * 3 + 2] = Pz[j];
    qf[s * 3 + 0] = Fx[j]; qf[s * 3 + 1] = Fy[j]; qf[s * 3 + 2] = Fz[j];
  }
}

// ---------------------------------------------------------------------------
// 2) Fourier embed (f16 output): [p(3) | sin(24) | cos(24) | feat(3) | 0 pad]
// ---------------------------------------------------------------------------
__global__ __launch_bounds__(256) void embed_k(
    const float* __restrict__ pts, const float* __restrict__ fts,
    _Float16* __restrict__ X, int R) {
  const int i = blockIdx.x * 256 + threadIdx.x;
  if (i >= R * KPAD) return;
  const int r = i >> 6, c = i & 63;
  float v;
  if (c < 3) {
    v = pts[(size_t)r * 3 + c];
  } else if (c < 27) {
    const int k = c - 3, ax = k >> 3, fr = k & 7;
    v = sinf(pts[(size_t)r * 3 + ax] * (float)(1 << fr));
  } else if (c < 51) {
    const int k = c - 27, ax = k >> 3, fr = k & 7;
    v = cosf(pts[(size_t)r * 3 + ax] * (float)(1 << fr));
  } else if (c < 54) {
    v = fts[(size_t)r * 3 + (c - 51)];
  } else {
    v = 0.0f;
  }
  X[(size_t)r * KPAD + c] = (_Float16)v;
}

__global__ __launch_bounds__(256) void pad_w_in_k(
    const float* __restrict__ w_in, _Float16* __restrict__ wpad) {
  const int i = blockIdx.x * 256 + threadIdx.x;
  if (i >= KPAD * WIDTH) return;
  const int r = i >> 10;
  wpad[i] = (_Float16)((r < 54) ? w_in[(size_t)r * WIDTH + (i & 1023)] : 0.0f);
}

// f32 -> f16 weight conversion (n multiple of 4)
__global__ __launch_bounds__(256) void cvt_f16_k(
    const float* __restrict__ x, _Float16* __restrict__ y, int n4) {
  const int i = blockIdx.x * 256 + threadIdx.x;
  if (i >= n4) return;
  const float4 v = ((const float4*)x)[i];
  _Float16* yp = y + (size_t)i * 4;
  yp[0] = (_Float16)v.x; yp[1] = (_Float16)v.y;
  yp[2] = (_Float16)v.z; yp[3] = (_Float16)v.w;
}

// ---------------------------------------------------------------------------
// 3) Row LayerNorm (eps 1e-5), one block per row; f32 and/or f16 outputs
// ---------------------------------------------------------------------------
__global__ __launch_bounds__(256) void layernorm_k(
    const float* __restrict__ x, const float* __restrict__ w,
    const float* __restrict__ bb, float* __restrict__ yf,
    _Float16* __restrict__ yh, int D) {
  __shared__ float s1[256], s2[256];
  const int row = blockIdx.x, tid = threadIdx.x;
  const float* xr = x + (size_t)row * D;
  float a = 0.0f, b2 = 0.0f;
  for (int i = tid; i < D; i += 256) { const float v = xr[i]; a += v; b2 += v * v; }
  s1[tid] = a; s2[tid] = b2;
  __syncthreads();
  for (int off = 128; off > 0; off >>= 1) {
    if (tid < off) { s1[tid] += s1[tid + off]; s2[tid] += s2[tid + off]; }
    __syncthreads();
  }
  const float mean = s1[0] / (float)D;
  const float var  = s2[0] / (float)D - mean * mean;
  const float rstd = rsqrtf(var + 1e-5f);
  for (int i = tid; i < D; i += 256) {
    const float v = (xr[i] - mean) * rstd * w[i] + bb[i];
    if (yf) yf[(size_t)row * D + i] = v;
    if (yh) yh[(size_t)row * D + i] = (_Float16)v;
  }
}

// ---------------------------------------------------------------------------
// 4) WMMA GEMM: C = act(A[MxK]h @ B[KxN]h + bias) + residual
//    A/B f16 in global; async b128 staging to LDS; f32 and/or f16 outputs.
//    M, N multiples of 128; K multiple of 32.
// ---------------------------------------------------------------------------
#define BM 128
#define BN 128
#define BK 32
__global__ __launch_bounds__(256) void gemm_wmma_k(
    const _Float16* __restrict__ A, const _Float16* __restrict__ Bw,
    float* __restrict__ Cf, _Float16* __restrict__ Ch,
    const float* __restrict__ bias, const float* __restrict__ res,
    int M, int N, int K, int act) {
  __shared__ _Float16 As[BM * BK];      // row-major, K contiguous
  __shared__ _Float16 Bs[BN * BK];      // N-major (transposed), K contiguous
  const int tid  = threadIdx.x;
  const int lane = tid & 31;
  const int wave = tid >> 5;            // 0..7
  const int wr   = wave >> 1;           // rows wr*32 + {0,16}
  const int wc   = wave & 1;            // cols wc*64 + {0,16,32,48}
  const int bm = blockIdx.y * BM;
  const int bn = blockIdx.x * BN;

  v8f acc[2][4];
#pragma unroll
  for (int i = 0; i < 2; ++i)
#pragma unroll
    for (int j = 0; j < 4; ++j) acc[i][j] = (v8f){};

  for (int k0 = 0; k0 < K; k0 += BK) {
    __syncthreads();
    // ---- B tile: b128 load + transpose scatter into N-major LDS ----
#pragma unroll
    for (int j = 0; j < 2; ++j) {
      const int chunk = tid + j * 256;          // 512 chunks of 8 halves
      const int kk = chunk >> 4, n8 = (chunk & 15) << 3;
      const v8h bvec = *(const v8h*)&Bw[(size_t)(k0 + kk) * N + bn + n8];
#pragma unroll
      for (int e = 0; e < 8; ++e) Bs[(n8 + e) * BK + kk] = bvec[e];
    }
    // ---- A tile: direct 16B copies (async to LDS when available) ----
#pragma unroll
    for (int j = 0; j < 2; ++j) {
      const int chunk = tid + j * 256;
      const int r = chunk >> 2, c8 = (chunk & 3) << 3;
      const _Float16* gp = &A[(size_t)(bm + r) * K + k0 + c8];
      _Float16* lp = &As[r * BK + c8];
#if HAS_ASYNC_LDS
      async_copy16(gp, lp);
#else
      *(v8h*)lp = *(const v8h*)gp;
#endif
    }
    // ---- prefetch next k-step tiles into L2 ----
    if (k0 + BK < K) {
      __builtin_prefetch(&A[(size_t)(bm + (tid >> 2)) * K + k0 + BK], 0, 1);
      __builtin_prefetch(&Bw[(size_t)(k0 + BK + (tid >> 4)) * N + bn + ((tid & 15) << 3)], 0, 1);
    }
#if HAS_ASYNC_LDS
    async_wait0();
#endif
    __syncthreads();

    v16h af[2], bf[4];
#pragma unroll
    for (int t = 0; t < 2; ++t) af[t] = load_frag(As, wr * 32 + t * 16, BK, lane);
#pragma unroll
    for (int t = 0; t < 4; ++t) bf[t] = load_frag(Bs, wc * 64 + t * 16, BK, lane);
#pragma unroll
    for (int tm = 0; tm < 2; ++tm)
#pragma unroll
      for (int tn = 0; tn < 4; ++tn)
        acc[tm][tn] = wmma_f16(af[tm], bf[tn], acc[tm][tn]);
  }

  // Epilogue: lane 0-15 -> col=lane, rows 0..7; lane 16-31 -> rows 8..15
  const int lr = lane & 15;
  const int hi = (lane >> 4) << 3;
#pragma unroll
  for (int tm = 0; tm < 2; ++tm) {
#pragma unroll
    for (int tn = 0; tn < 4; ++tn) {
      const int col = bn + wc * 64 + tn * 16 + lr;
      const float bv = bias ? bias[col] : 0.0f;
#pragma unroll
      for (int r = 0; r < 8; ++r) {
        const int row = bm + wr * 32 + tm * 16 + hi + r;
        float v = acc[tm][tn][r] + bv;
        if (act == 1) v = gelu_exact(v);
        if (res) v += res[(size_t)row * N + col];
        if (Cf) Cf[(size_t)row * N + col] = v;
        if (Ch) Ch[(size_t)row * N + col] = (_Float16)v;
      }
    }
  }
}

// ---------------------------------------------------------------------------
// 5) Flash attention (f16 I/O) with fused qk-head-layernorm.
//    grid = (Sq/64, HEADS, BATCH); block = 256 (8 waves); 64 q-rows / block.
// ---------------------------------------------------------------------------
__global__ __launch_bounds__(256) void attn_wmma_k(
    const _Float16* __restrict__ Q, const _Float16* __restrict__ Kd,
    const _Float16* __restrict__ Vd, _Float16* __restrict__ O,
    const float* __restrict__ qw, const float* __restrict__ qb,
    const float* __restrict__ kw, const float* __restrict__ kbi,
    int Sq, int Sk, int ldq, int ldkv) {
  __shared__ float    Sf[64][64];      // raw staging + scores
  __shared__ _Float16 Qs[64 * 64];
  __shared__ _Float16 Ks[64 * 64];
  __shared__ _Float16 Vs[64 * 64];     // transposed: Vs[d][key]
  __shared__ _Float16 Ps[64 * 64];
  __shared__ float mrow[64], lrow[64], arow[64];

  const int b = blockIdx.z, h = blockIdx.y, q0 = blockIdx.x * 64;
  const int tid = threadIdx.x, lane = tid & 31, wave = tid >> 5;
  const int wr  = wave >> 1;            // row tile (16 rows at wr*16)
  const int wcb = (wave & 1) * 32;      // col base (tiles wcb, wcb+16)
  const _Float16* Qh = Q  + (size_t)b * Sq * ldq  + (size_t)h * HD;
  const _Float16* Kh = Kd + (size_t)b * Sk * ldkv + (size_t)h * HD;
  const _Float16* Vh = Vd + (size_t)b * Sk * ldkv + (size_t)h * HD;

  // Stage + layernorm Q tile
  for (int i = tid; i < 64 * 64; i += 256) {
    const int r = i >> 6, c = i & 63;
    Sf[r][c] = (float)Qh[(size_t)(q0 + r) * ldq + c];
  }
  __syncthreads();
  if (tid < 64) {
    float s = 0.0f, s2 = 0.0f;
    for (int d = 0; d < HD; ++d) { const float v = Sf[tid][d]; s += v; s2 += v * v; }
    const float mean = s * (1.0f / HD);
    const float rstd = rsqrtf(s2 * (1.0f / HD) - mean * mean + 1e-5f);
    for (int d = 0; d < HD; ++d)
      Qs[tid * 64 + d] = (_Float16)((Sf[tid][d] - mean) * rstd * qw[d] + qb[d]);
    mrow[tid] = -1e30f;
    lrow[tid] = 0.0f;
  }
  __syncthreads();

  v8f oa[2] = {(v8f){}, (v8f){}};
  const int lr = lane & 15;
  const int hi = (lane >> 4) << 3;

  for (int kb0 = 0; kb0 < Sk; kb0 += 64) {
    __syncthreads();   // protect Sf/Ks/Vs/Ps from prior-iteration readers
    for (int i = tid; i < 64 * 64; i += 256) {
      const int r = i >> 6, c = i & 63;
      const _Float16 kv = Kh[(size_t)(kb0 + r) * ldkv + c];
      Sf[r][c] = (float)kv;
      Vs[c * 64 + r] = Vh[(size_t)(kb0 + r) * ldkv + c];
    }
    __syncthreads();
    if (tid < 64) {  // layernorm K rows
      float s = 0.0f, s2 = 0.0f;
      for (int d = 0; d < HD; ++d) { const float v = Sf[tid][d]; s += v; s2 += v * v; }
      const float mean = s * (1.0f / HD);
      const float rstd = rsqrtf(s2 * (1.0f / HD) - mean * mean + 1e-5f);
      for (int d = 0; d < HD; ++d)
        Ks[tid * 64 + d] = (_Float16)((Sf[tid][d] - mean) * rstd * kw[d] + kbi[d]);
    }
    __syncthreads();

    // S = (Q @ K^T) * 1/sqrt(64); each wave: 16x32 of the 64x64 score tile
    v8f sa[2] = {(v8f){}, (v8f){}};
#pragma unroll
    for (int kk = 0; kk < 64; kk += 32) {
      const v16h aq = load_frag(Qs + kk, wr * 16, 64, lane);
#pragma unroll
      for (int t = 0; t < 2; ++t) {
        const v16h bk = load_frag(Ks + kk, wcb + t * 16, 64, lane);
        sa[t] = wmma_f16(aq, bk, sa[t]);
      }
    }
#pragma unroll
    for (int t = 0; t < 2; ++t)
#pragma unroll
      for (int r = 0; r < 8; ++r)
        Sf[wr * 16 + hi + r][wcb + t * 16 + lr] = sa[t][r] * 0.125f;
    __syncthreads();

    // Online softmax: one thread per query row
    if (tid < 64) {
      const float mold = mrow[tid];
      float mx = mold;
      for (int j = 0; j < 64; ++j) mx = fmaxf(mx, Sf[tid][j]);
      const float alpha = expf(mold - mx);
      float ssum = 0.0f;
      for (int j = 0; j < 64; ++j) {
        const float p = expf(Sf[tid][j] - mx);
        Ps[tid * 64 + j] = (_Float16)p;
        ssum += p;
      }
      lrow[tid] = alpha * lrow[tid] + ssum;
      mrow[tid] = mx;
      arow[tid] = alpha;
    }
    __syncthreads();

    // Rescale O accumulators, then O += P @ V
#pragma unroll
    for (int r = 0; r < 8; ++r) {
      const float al = arow[wr * 16 + hi + r];
      oa[0][r] *= al;
      oa[1][r] *= al;
    }
#pragma unroll
    for (int kk = 0; kk < 64; kk += 32) {
      const v16h ap = load_frag(Ps + kk, wr * 16, 64, lane);
#pragma unroll
      for (int t = 0; t < 2; ++t) {
        const v16h bv = load_frag(Vs + kk, wcb + t * 16, 64, lane);
        oa[t] = wmma_f16(ap, bv, oa[t]);
      }
    }
  }

  // Write O (f16, feeds next GEMM as A operand)
#pragma unroll
  for (int t = 0; t < 2; ++t) {
    const int col = h * HD + wcb + t * 16 + lr;
#pragma unroll
    for (int r = 0; r < 8; ++r) {
      const int row = wr * 16 + hi + r;
      O[((size_t)b * Sq + q0 + row) * WIDTH + col] = (_Float16)(oa[t][r] / lrow[row]);
    }
  }
}

// ---------------------------------------------------------------------------
// Host orchestration
// ---------------------------------------------------------------------------
enum {
  IN_PC = 0, IN_FEATS, IN_PERM_RAND, IN_PERM_SHARP,
  P_W_IN, P_B_IN,
  CA_LN1W, CA_LN1B, CA_LN2W, CA_LN2B, CA_LN3W, CA_LN3B,
  CA_WQ, CA_WKV, CA_QLNW, CA_QLNB, CA_KLNW, CA_KLNB,
  CA_WO, CA_BO, CA_MW1, CA_MB1, CA_MW2, CA_MB2,
  SA_LN1W, SA_LN1B, SA_LN2W, SA_LN2B, SA_WQKV,
  SA_QLNW, SA_QLNB, SA_KLNW, SA_KLNB,
  SA_WO, SA_BO, SA_MW1, SA_MB1, SA_MW2, SA_MB2,
  P_POSTW, P_POSTB
};

static inline void launch_gemm(const _Float16* A, const _Float16* B,
                               float* Cf, _Float16* Ch,
                               const float* bias, const float* res,
                               int M, int N, int K, int act, hipStream_t s) {
  dim3 grid(N / BN, M / BM);
  gemm_wmma_k<<<grid, 256, 0, s>>>(A, B, Cf, Ch, bias, res, M, N, K, act);
}

static inline void launch_ln(const float* x, const float* w, const float* b,
                             float* yf, _Float16* yh, int rows, hipStream_t s) {
  layernorm_k<<<rows, 256, 0, s>>>(x, w, b, yf, yh, WIDTH);
}

static inline void launch_cvt(const float* x, _Float16* y, size_t n, hipStream_t s) {
  const int n4 = (int)(n / 4);
  cvt_f16_k<<<(n4 + 255) / 256, 256, 0, s>>>(x, y, n4);
}

extern "C" void kernel_launch(void* const* d_in, const int* in_sizes, int n_in,
                              void* d_out, int out_size, void* d_ws, size_t ws_size,
                              hipStream_t stream) {
  (void)in_sizes; (void)n_in; (void)out_size; (void)ws_size;

  const float* pc     = (const float*)d_in[IN_PC];
  const float* feats  = (const float*)d_in[IN_FEATS];
  const int*   permR  = (const int*)d_in[IN_PERM_RAND];
  const int*   permS  = (const int*)d_in[IN_PERM_SHARP];
  const float* w_in   = (const float*)d_in[P_W_IN];
  const float* b_in   = (const float*)d_in[P_B_IN];
  float* out          = (float*)d_out;

  // Workspace carve-up (256B aligned)
  char* base = (char*)d_ws;
  size_t off = 0;
  auto allocB = [&](size_t bytes) {
    void* p = base + off;
    off += (bytes + 255) & ~(size_t)255;
    return p;
  };
  // f32 buffers
  float* inpPts = (float*)allocB((size_t)BATCH * N_DATA * 3 * 4);
  float* qPts   = (float*)allocB((size_t)BATCH * N_QUERY * 3 * 4);
  float* inpFts = (float*)allocB((size_t)BATCH * N_DATA * 3 * 4);
  float* qFts   = (float*)allocB((size_t)BATCH * N_QUERY * 3 * 4);
  float* D0     = (float*)allocB((size_t)M_D * WIDTH * 4);
  float* X      = (float*)allocB((size_t)M_Q * WIDTH * 4);
  // f16 activations
  _Float16* Xd   = (_Float16*)allocB((size_t)M_D * KPAD * 2);
  _Float16* Xq   = (_Float16*)allocB((size_t)M_Q * KPAD * 2);
  _Float16* wpad = (_Float16*)allocB((size_t)KPAD * WIDTH * 2);
  _Float16* XNh  = (_Float16*)allocB((size_t)M_Q * WIDTH * 2);
  _Float16* DNh  = (_Float16*)allocB((size_t)M_D * WIDTH * 2);
  _Float16* Qbh  = (_Float16*)allocB((size_t)M_Q * WIDTH * 2);
  _Float16* Obh  = (_Float16*)allocB((size_t)M_Q * WIDTH * 2);
  _Float16* KVh  = (_Float16*)allocB((size_t)M_D * 2 * WIDTH * 2);  // also QKV
  _Float16* QKVh = KVh;                                             // 2048x3072 fits
  _Float16* Hh   = (_Float16*)allocB((size_t)M_Q * 4 * WIDTH * 2);
  // f16 weight mirrors
  _Float16* wq_h   = (_Float16*)allocB((size_t)WIDTH * WIDTH * 2);
  _Float16* wkv_h  = (_Float16*)allocB((size_t)WIDTH * 2 * WIDTH * 2);
  _Float16* wo_h   = (_Float16*)allocB((size_t)WIDTH * WIDTH * 2);
  _Float16* mw1_h  = (_Float16*)allocB((size_t)WIDTH * 4 * WIDTH * 2);
  _Float16* mw2_h  = (_Float16*)allocB((size_t)4 * WIDTH * WIDTH * 2);
  _Float16* swqkv_h = (_Float16*)allocB((size_t)WIDTH * 3 * WIDTH * 2);
  _Float16* swo_h   = (_Float16*)allocB((size_t)WIDTH * WIDTH * 2);
  _Float16* smw1_h  = (_Float16*)allocB((size_t)WIDTH * 4 * WIDTH * 2);
  _Float16* smw2_h  = (_Float16*)allocB((size_t)4 * WIDTH * WIDTH * 2);

  // --- 1. FPS subsample + gathers ---
  fps_gather_k<<<BATCH * 2, 256, 0, stream>>>(pc, feats, permR, permS,
                                              inpPts, qPts, inpFts, qFts);

  // --- 2. Embeds + weight conversion ---
  embed_k<<<(M_D * KPAD + 255) / 256, 256, 0, stream>>>(inpPts, inpFts, Xd, M_D);
  embed_k<<<(M_Q * KPAD + 255) / 256, 256, 0, stream>>>(qPts, qFts, Xq, M_Q);
  pad_w_in_k<<<(KPAD * WIDTH + 255) / 256, 256, 0, stream>>>(w_in, wpad);
  launch_cvt((const float*)d_in[CA_WQ],  wq_h,  (size_t)WIDTH * WIDTH, stream);
  launch_cvt((const float*)d_in[CA_WKV], wkv_h, (size_t)WIDTH * 2 * WIDTH, stream);
  launch_cvt((const float*)d_in[CA_WO],  wo_h,  (size_t)WIDTH * WIDTH, stream);
  launch_cvt((const float*)d_in[CA_MW1], mw1_h, (size_t)WIDTH * 4 * WIDTH, stream);
  launch_cvt((const float*)d_in[CA_MW2], mw2_h, (size_t)4 * WIDTH * WIDTH, stream);

  // --- 3. Input projections ---
  launch_gemm(Xd, wpad, D0, nullptr, b_in, nullptr, M_D, WIDTH, KPAD, 0, stream);
  launch_gemm(Xq, wpad, X,  nullptr, b_in, nullptr, M_Q, WIDTH, KPAD, 0, stream);

  // --- 4. Cross-attention block ---
  launch_ln(X,  (const float*)d_in[CA_LN1W], (const float*)d_in[CA_LN1B], nullptr, XNh, M_Q, stream);
  launch_ln(D0, (const float*)d_in[CA_LN2W], (const float*)d_in[CA_LN2B], nullptr, DNh, M_D, stream);
  launch_gemm(XNh, wq_h,  nullptr, Qbh, nullptr, nullptr, M_Q, WIDTH,     WIDTH, 0, stream);
  launch_gemm(DNh, wkv_h, nullptr, KVh, nullptr, nullptr, M_D, 2 * WIDTH, WIDTH, 0, stream);
  {
    dim3 grid(N_QUERY / 64, HEADS, BATCH);
    attn_wmma_k<<<grid, 256, 0, stream>>>(
        Qbh, KVh, KVh + WIDTH, Obh,
        (const float*)d_in[CA_QLNW], (const float*)d_in[CA_QLNB],
        (const float*)d_in[CA_KLNW], (const float*)d_in[CA_KLNB],
        N_QUERY, N_DATA, WIDTH, 2 * WIDTH);
  }
  launch_gemm(Obh, wo_h, X, nullptr, (const float*)d_in[CA_BO], X,
              M_Q, WIDTH, WIDTH, 0, stream);
  launch_ln(X, (const float*)d_in[CA_LN3W], (const float*)d_in[CA_LN3B], nullptr, XNh, M_Q, stream);
  launch_gemm(XNh, mw1_h, nullptr, Hh, (const float*)d_in[CA_MB1], nullptr,
              M_Q, 4 * WIDTH, WIDTH, 1, stream);
  launch_gemm(Hh, mw2_h, X, nullptr, (const float*)d_in[CA_MB2], X,
              M_Q, WIDTH, 4 * WIDTH, 0, stream);

  // --- 5. Self-attention layers ---
  for (int i = 0; i < 4; ++i) {
    const float* ln1w = (const float*)d_in[SA_LN1W] + (size_t)i * WIDTH;
    const float* ln1b = (const float*)d_in[SA_LN1B] + (size_t)i * WIDTH;
    const float* ln2w = (const float*)d_in[SA_LN2W] + (size_t)i * WIDTH;
    const float* ln2b = (const float*)d_in[SA_LN2B] + (size_t)i * WIDTH;
    const float* wqkv = (const float*)d_in[SA_WQKV] + (size_t)i * WIDTH * 3 * WIDTH;
    const float* qlnw = (const float*)d_in[SA_QLNW] + (size_t)i * HD;
    const float* qlnb = (const float*)d_in[SA_QLNB] + (size_t)i * HD;
    const float* klnw = (const float*)d_in[SA_KLNW] + (size_t)i * HD;
    const float* klnb = (const float*)d_in[SA_KLNB] + (size_t)i * HD;
    const float* wo   = (const float*)d_in[SA_WO]   + (size_t)i * WIDTH * WIDTH;
    const float* bo   = (const float*)d_in[SA_BO]   + (size_t)i * WIDTH;
    const float* mw1  = (const float*)d_in[SA_MW1]  + (size_t)i * WIDTH * 4 * WIDTH;
    const float* mb1  = (const float*)d_in[SA_MB1]  + (size_t)i * 4 * WIDTH;
    const float* mw2  = (const float*)d_in[SA_MW2]  + (size_t)i * 4 * WIDTH * WIDTH;
    const float* mb2  = (const float*)d_in[SA_MB2]  + (size_t)i * WIDTH;

    launch_cvt(wqkv, swqkv_h, (size_t)WIDTH * 3 * WIDTH, stream);
    launch_cvt(wo,   swo_h,   (size_t)WIDTH * WIDTH, stream);
    launch_cvt(mw1,  smw1_h,  (size_t)WIDTH * 4 * WIDTH, stream);
    launch_cvt(mw2,  smw2_h,  (size_t)4 * WIDTH * WIDTH, stream);

    launch_ln(X, ln1w, ln1b, nullptr, XNh, M_Q, stream);
    launch_gemm(XNh, swqkv_h, nullptr, QKVh, nullptr, nullptr, M_Q, 3 * WIDTH, WIDTH, 0, stream);
    {
      dim3 grid(N_QUERY / 64, HEADS, BATCH);
      attn_wmma_k<<<grid, 256, 0, stream>>>(
          QKVh, QKVh + WIDTH, QKVh + 2 * WIDTH, Obh,
          qlnw, qlnb, klnw, klnb,
          N_QUERY, N_QUERY, 3 * WIDTH, 3 * WIDTH);
    }
    launch_gemm(Obh, swo_h, X, nullptr, bo, X, M_Q, WIDTH, WIDTH, 0, stream);
    launch_ln(X, ln2w, ln2b, nullptr, XNh, M_Q, stream);
    launch_gemm(XNh, smw1_h, nullptr, Hh, mb1, nullptr, M_Q, 4 * WIDTH, WIDTH, 1, stream);
    launch_gemm(Hh, smw2_h, X, nullptr, mb2, X, M_Q, WIDTH, 4 * WIDTH, 0, stream);
  }

  // --- 6. Final layernorm -> output ---
  launch_ln(X, (const float*)d_in[P_POSTW], (const float*)d_in[P_POSTB], out, nullptr, M_Q, stream);
}
